// Transformer_75711683493977
// MI455X (gfx1250) — compile-verified, matrix operand order checked by reading, not run on hardware
//
#include <hip/hip_runtime.h>
#include <cstdint>

// ---------------------------------------------------------------------------
// Types for CDNA5 WMMA (wave32): v_wmma_f32_16x16x32_bf16
// ---------------------------------------------------------------------------
typedef __attribute__((ext_vector_type(16))) __bf16 v16bf;
typedef __attribute__((ext_vector_type(8)))  __bf16 v8bf;
typedef __attribute__((ext_vector_type(8)))  float  v8f;
typedef __attribute__((ext_vector_type(4)))  unsigned int v4u;
typedef __attribute__((ext_vector_type(8)))  int    v8i;
typedef __attribute__((ext_vector_type(4)))  int    v4i;

__device__ __forceinline__ v8f wmma_bf16(v16bf a, v16bf b, v8f c) {
    return __builtin_amdgcn_wmma_f32_16x16x32_bf16(
        false, a, false, b, (short)0, c, false, false);
}

// A-fragment (16x32, M x K) / symmetric B-fragment loader from row-major bf16.
// Lane l: r = l&15, hi = l>>4 -> K runs [hi*8,+8) and [16+hi*8,+8).
__device__ __forceinline__ v16bf load_frag_rm(const __bf16* base, int stride, int lane) {
    int r = lane & 15, hi = lane >> 4;
    const __bf16* p = base + (size_t)r * stride + hi * 8;
    v8bf a = *reinterpret_cast<const v8bf*>(p);
    v8bf b = *reinterpret_cast<const v8bf*>(p + 16);
    return __builtin_shufflevector(a, b, 0,1,2,3,4,5,6,7,8,9,10,11,12,13,14,15);
}

// ---------------------------------------------------------------------------
// Tensor Data Mover: 2D tile (rows x 32 bf16) global -> LDS, row-major.
// D# packed per CDNA5 ISA 8.3/8.4.
// ---------------------------------------------------------------------------
#if __has_builtin(__builtin_amdgcn_tensor_load_to_lds)
#define HAVE_TDM 1
#endif

__device__ __forceinline__ void tdm_load_tile(const __bf16* gbase, unsigned rows,
                                              unsigned K, void* ldsdst) {
#ifdef HAVE_TDM
    unsigned long long ga = (unsigned long long)(uintptr_t)gbase;
    unsigned lds = (unsigned)(uintptr_t)ldsdst;
    v4u g0;
    g0.x = 1u;                                   // count=1, user mode
    g0.y = lds;                                  // lds_addr (bytes)
    g0.z = (unsigned)ga;                         // global_addr[31:0]
    g0.w = (unsigned)((ga >> 32) & 0x01FFFFFFull) | 0x80000000u;  // addr[56:32] | type=2
    v8i g1;
    g1[0] = 1 << 16;                             // data_size=1 -> 2 bytes, wg_mask=0
    g1[1] = (int)((K & 0xFFFFu) << 16);          // tensor_dim0[15:0]
    g1[2] = (int)(((K >> 16) & 0xFFFFu) | ((rows & 0xFFFFu) << 16));   // dim0 hi | dim1 lo
    g1[3] = (int)(((rows >> 16) & 0xFFFFu) | (32u << 16));             // dim1 hi | tile_dim0
    g1[4] = (int)(rows & 0xFFFFu);               // tile_dim1, tile_dim2=0
    g1[5] = (int)K;                              // tensor_dim0_stride[31:0]
    g1[6] = 0;
    g1[7] = 0;
    v4i z4 = {0, 0, 0, 0};
#if __clang_major__ >= 23
    v8i z8 = {0, 0, 0, 0, 0, 0, 0, 0};
    __builtin_amdgcn_tensor_load_to_lds(g0, g1, z4, z4, z8, 0);
#else
    __builtin_amdgcn_tensor_load_to_lds(g0, g1, z4, z4, 0);
#endif
#endif
}

__device__ __forceinline__ void wait_tensorcnt0() {
#if __has_builtin(__builtin_amdgcn_s_wait_tensorcnt)
    __builtin_amdgcn_s_wait_tensorcnt(0);
#else
    asm volatile("s_wait_tensorcnt 0" ::: "memory");
#endif
}

// ---------------------------------------------------------------------------
// Model constants
// ---------------------------------------------------------------------------
constexpr int Bc = 2, Tc = 2048, Dc = 512, Hc = 8, HWc = 64, HDc = 2048, Lc = 8, Vc = 32000;
constexpr int Rc = Bc * Tc;

// ---------------------------------------------------------------------------
// Embedding gather
// ---------------------------------------------------------------------------
__global__ __launch_bounds__(256) void gather_kernel(const int* __restrict__ tokens,
                                                     const float* __restrict__ embed,
                                                     float* __restrict__ x) {
    size_t idx = (size_t)blockIdx.x * 256 + threadIdx.x;
    int row = (int)(idx >> 9), col = (int)(idx & 511);
    x[idx] = embed[(size_t)tokens[row] * Dc + col];
}

// ---------------------------------------------------------------------------
// LayerNorm (d=512), one row per 256-thread block; writes f32 + bf16.
// ---------------------------------------------------------------------------
__global__ __launch_bounds__(256) void ln_kernel(const float* __restrict__ x,
                                                 const float* __restrict__ g,
                                                 const float* __restrict__ bb,
                                                 float* __restrict__ outf,
                                                 __bf16* __restrict__ outb) {
    __shared__ float sums[8], sqs[8];
    int row = blockIdx.x, tid = threadIdx.x;
    const float* xr = x + (size_t)row * Dc;
    float v0 = xr[tid], v1 = xr[tid + 256];
    float s = v0 + v1, q = v0 * v0 + v1 * v1;
#pragma unroll
    for (int m = 1; m < 32; m <<= 1) { s += __shfl_xor(s, m, 32); q += __shfl_xor(q, m, 32); }
    if ((tid & 31) == 0) { sums[tid >> 5] = s; sqs[tid >> 5] = q; }
    __syncthreads();
    float ts = 0.f, tq = 0.f;
#pragma unroll
    for (int i = 0; i < 8; ++i) { ts += sums[i]; tq += sqs[i]; }
    float mu = ts * (1.0f / Dc);
    float var = tq * (1.0f / Dc) - mu * mu;
    float rstd = rsqrtf(var + 1e-5f);
    float o0 = (v0 - mu) * rstd * g[tid] + bb[tid];
    float o1 = (v1 - mu) * rstd * g[tid + 256] + bb[tid + 256];
    size_t base = (size_t)row * Dc;
    outf[base + tid] = o0;        outb[base + tid] = (__bf16)o0;
    outf[base + tid + 256] = o1;  outb[base + tid + 256] = (__bf16)o1;
}

// ---------------------------------------------------------------------------
// Elementwise helpers
// ---------------------------------------------------------------------------
__global__ __launch_bounds__(256) void add_kernel(const float* __restrict__ a,
                                                  const float* __restrict__ b,
                                                  float* __restrict__ o) {
    size_t i = (size_t)blockIdx.x * 256 + threadIdx.x;
    o[i] = a[i] + b[i];
}

__global__ __launch_bounds__(256) void cvt_kernel(const float* __restrict__ in,
                                                  __bf16* __restrict__ out) {
    size_t i = (size_t)blockIdx.x * 256 + threadIdx.x;
    out[i] = (__bf16)in[i];
}

// ---------------------------------------------------------------------------
// TDM-fed, LDS-staged WMMA GEMM: C[M,N] = A[M,K] @ W[N,K]^T, all bf16 inputs.
// Block tile 128x256 (8 waves, wave tile 64x64 = 16 WMMAs per 32-K step).
// Wave 0 drives the TDM with double-buffered A/B tiles; everyone consumes
// fragments from LDS.
// MODE 0: outf=acc | 1: outb=bf16(acc) | 2: outf=resid+acc
// MODE 3: outb=bf16(silu(acc)) | 4: outb = bf16(acc) stored V-transposed
//         as Vt[b][col][t] with b=row>>11, t=row&2047 (stride Tc along keys)
// ---------------------------------------------------------------------------
template <int MODE>
__global__ __launch_bounds__(256) void gemm_kernel(const __bf16* __restrict__ A,
                                                   const __bf16* __restrict__ W,
                                                   const float* __restrict__ resid,
                                                   float* __restrict__ outf,
                                                   __bf16* __restrict__ outb,
                                                   int M, int N, int K) {
    __shared__ __align__(16) __bf16 sA[2][128 * 32];
    __shared__ __align__(16) __bf16 sB[2][256 * 32];

    int tid  = threadIdx.x;
    int lane = tid & 31;
    int wave = tid >> 5;                 // 0..7
    int wm = wave >> 2, wn = wave & 3;   // 2 x 4 wave grid
    int mblk = blockIdx.x * 128;
    int nblk = blockIdx.y * 256;

    const __bf16* Abase = A + (size_t)mblk * K;
    const __bf16* Wbase = W + (size_t)nblk * K;

    v8f acc[4][4];
#pragma unroll
    for (int i = 0; i < 4; ++i)
#pragma unroll
        for (int j = 0; j < 4; ++j) acc[i][j] = (v8f){};

#ifdef HAVE_TDM
    if (wave == 0) {
        tdm_load_tile(Abase, 128, (unsigned)K, &sA[0][0]);
        tdm_load_tile(Wbase, 256, (unsigned)K, &sB[0][0]);
        wait_tensorcnt0();
    }
    __syncthreads();
#else
    for (int c = tid; c < 128 * 4; c += 256) {
        int r = c >> 2, c8 = (c & 3) * 8;
        *reinterpret_cast<v8bf*>(&sA[0][r * 32 + c8]) =
            *reinterpret_cast<const v8bf*>(Abase + (size_t)r * K + c8);
    }
    for (int c = tid; c < 256 * 4; c += 256) {
        int r = c >> 2, c8 = (c & 3) * 8;
        *reinterpret_cast<v8bf*>(&sB[0][r * 32 + c8]) =
            *reinterpret_cast<const v8bf*>(Wbase + (size_t)r * K + c8);
    }
    __syncthreads();
#endif

    int cur = 0;
    for (int k0 = 0; k0 < K; k0 += 32) {
        int nxt = cur ^ 1;
#ifdef HAVE_TDM
        if (wave == 0 && k0 + 32 < K) {
            tdm_load_tile(Abase + (k0 + 32), 128, (unsigned)K, &sA[nxt][0]);
            tdm_load_tile(Wbase + (k0 + 32), 256, (unsigned)K, &sB[nxt][0]);
        }
#endif
        v16bf afrag[4];
#pragma unroll
        for (int i = 0; i < 4; ++i)
            afrag[i] = load_frag_rm(&sA[cur][(wm * 64 + 16 * i) * 32], 32, lane);
#pragma unroll
        for (int j = 0; j < 4; ++j) {
            v16bf bfrag = load_frag_rm(&sB[cur][(wn * 64 + 16 * j) * 32], 32, lane);
#pragma unroll
            for (int i = 0; i < 4; ++i)
                acc[i][j] = wmma_bf16(afrag[i], bfrag, acc[i][j]);
        }
#ifdef HAVE_TDM
        if (wave == 0) wait_tensorcnt0();
        __syncthreads();
#else
        __syncthreads();
        if (k0 + 32 < K) {
            for (int c = tid; c < 128 * 4; c += 256) {
                int r = c >> 2, c8 = (c & 3) * 8;
                *reinterpret_cast<v8bf*>(&sA[nxt][r * 32 + c8]) =
                    *reinterpret_cast<const v8bf*>(Abase + (size_t)r * K + k0 + 32 + c8);
            }
            for (int c = tid; c < 256 * 4; c += 256) {
                int r = c >> 2, c8 = (c & 3) * 8;
                *reinterpret_cast<v8bf*>(&sB[nxt][r * 32 + c8]) =
                    *reinterpret_cast<const v8bf*>(Wbase + (size_t)r * K + k0 + 32 + c8);
            }
            __syncthreads();
        }
#endif
        cur = nxt;
    }

    int ln = lane & 15, hi = lane >> 4;
    int m0 = mblk + wm * 64;
    int n0 = nblk + wn * 64;
#pragma unroll
    for (int i = 0; i < 4; ++i) {
#pragma unroll
        for (int j = 0; j < 4; ++j) {
#pragma unroll
            for (int e = 0; e < 8; ++e) {
                int mr = m0 + i * 16 + e + 8 * hi;
                int nc = n0 + j * 16 + ln;
                size_t idx = (size_t)mr * N + nc;
                float v = acc[i][j][e];
                if constexpr (MODE == 0) {
                    outf[idx] = v;
                } else if constexpr (MODE == 1) {
                    outb[idx] = (__bf16)v;
                } else if constexpr (MODE == 2) {
                    outf[idx] = resid[idx] + v;
                } else if constexpr (MODE == 3) {
                    float sv = v / (1.0f + __expf(-v));
                    outb[idx] = (__bf16)sv;
                } else {  // MODE == 4: V stored transposed per (batch, feature)
                    int bb2 = mr >> 11;          // row / Tc
                    int tt  = mr & (Tc - 1);     // row % Tc
                    size_t tidx = (size_t)bb2 * Dc * Tc + (size_t)nc * Tc + tt;
                    outb[tidx] = (__bf16)v;
                }
            }
        }
    }
}

// ---------------------------------------------------------------------------
// Flash attention with ALiBi + causal mask.
// One wave per (batch, head, 16-query tile); 64-key chunks.
// Phases per chunk: batch-load 8 K-fragments -> 8 S-WMMAs -> batch-load 8
// V^T-fragments (latency hidden under softmax) -> online softmax (ALiBi bias
// precomputed; mask == bias>0, no integer ops) -> P through LDS -> 8 PV-WMMAs.
// ---------------------------------------------------------------------------
__global__ __launch_bounds__(32) void attn_kernel(const __bf16* __restrict__ Q,
                                                  const __bf16* __restrict__ Kmat,
                                                  const __bf16* __restrict__ Vt,
                                                  float* __restrict__ O) {
    __shared__ __align__(16) __bf16 pbuf[16 * 64];

    constexpr int QT = Tc / 16;
    int bid = blockIdx.x;
    int qt = bid % QT;
    int h  = (bid / QT) % Hc;
    int bo = bid / (QT * Hc);
    int lane = threadIdx.x;
    int ln = lane & 15, hi = lane >> 4;

    const float slope = exp2f(-(float)(h + 1));   // 2^(-i*8/H), H=8
    const float scale = 0.125f;                   // 1/sqrt(64)
    const int qrow0 = qt * 16;

    const __bf16* qbase = Q + ((size_t)(bo * Tc + qrow0)) * Dc + h * HWc;
    v16bf aq0 = load_frag_rm(qbase, Dc, lane);
    v16bf aq1 = load_frag_rm(qbase + 32, Dc, lane);

    const __bf16* vtb = Vt + (size_t)bo * Dc * Tc + (size_t)h * HWc * Tc;

    // Chunk-invariant ALiBi term: nbias[n][i] = slope * ((n*16+ln) - qi).
    // Full bias at chunk kc is slope*kc + nbias; causal mask is bias > 0.
    float nbias[4][8];
#pragma unroll
    for (int i = 0; i < 8; ++i) {
        int qi = qrow0 + i + 8 * hi;
#pragma unroll
        for (int n = 0; n < 4; ++n)
            nbias[n][i] = slope * (float)((n * 16 + ln) - qi);
    }

    v8f acc[4];
#pragma unroll
    for (int j = 0; j < 4; ++j) acc[j] = (v8f){};
    float rmax[8], rsum[8];
#pragma unroll
    for (int i = 0; i < 8; ++i) { rmax[i] = -__builtin_inff(); rsum[i] = 0.f; }

    const int nchunks = (qrow0 + 16 + 63) >> 6;   // 64-key causal coverage
    for (int c = 0; c < nchunks; ++c) {
        int kc = c * 64;
        const float kcb = slope * (float)kc;

        // Batch-load all 8 K-fragments, then run the 8 S-WMMAs.
        const __bf16* kb = Kmat + ((size_t)(bo * Tc + kc)) * Dc + h * HWc;
        v16bf bk[8];
#pragma unroll
        for (int n = 0; n < 4; ++n) {
            bk[2 * n]     = load_frag_rm(kb + (size_t)(n * 16) * Dc, Dc, lane);
            bk[2 * n + 1] = load_frag_rm(kb + (size_t)(n * 16) * Dc + 32, Dc, lane);
        }
        v8f s[4];
#pragma unroll
        for (int n = 0; n < 4; ++n) {
            v8f t = {};
            t = wmma_bf16(aq0, bk[2 * n], t);
            t = wmma_bf16(aq1, bk[2 * n + 1], t);
            s[n] = t;
        }

        // Issue V^T fragment loads now; their latency hides under the softmax.
        v16bf bv[8];
#pragma unroll
        for (int j = 0; j < 4; ++j) {
            const __bf16* vb = vtb + (size_t)(j * 16) * Tc + kc;
            bv[2 * j]     = load_frag_rm(vb, Tc, lane);       // keys kc..kc+31
            bv[2 * j + 1] = load_frag_rm(vb + 32, Tc, lane);  // keys kc+32..kc+63
        }

        // Online softmax: bias = kcb + nbias; mask where bias > 0 (kj > qi).
        float p[4][8];
#pragma unroll
        for (int i = 0; i < 8; ++i) {
            float vv[4];
            float cm = -__builtin_inff();
#pragma unroll
            for (int n = 0; n < 4; ++n) {
                float bt = kcb + nbias[n][i];
                float val = s[n][i] * scale + bt;
                if (bt > 0.f) val = -__builtin_inff();
                vv[n] = val;
                cm = fmaxf(cm, val);
            }
#pragma unroll
            for (int m = 1; m < 16; m <<= 1) cm = fmaxf(cm, __shfl_xor(cm, m, 32));
            float nm = fmaxf(rmax[i], cm);
            float f  = __expf(rmax[i] - nm);
            float cs = 0.f;
#pragma unroll
            for (int n = 0; n < 4; ++n) {
                float e = __expf(vv[n] - nm);
                p[n][i] = e;
                cs += e;
            }
#pragma unroll
            for (int m = 1; m < 16; m <<= 1) cs += __shfl_xor(cs, m, 32);
            rsum[i] = rsum[i] * f + cs;
            rmax[i] = nm;
#pragma unroll
            for (int j = 0; j < 4; ++j) acc[j][i] *= f;
        }

        // P (D-layout) -> A-fragments via LDS
#pragma unroll
        for (int i = 0; i < 8; ++i) {
            int row = i + 8 * hi;
#pragma unroll
            for (int n = 0; n < 4; ++n)
                pbuf[row * 64 + n * 16 + ln] = (__bf16)p[n][i];
        }
        asm volatile("s_wait_dscnt 0" ::: "memory");
        v16bf ap0 = load_frag_rm(pbuf, 64, lane);       // keys kc..kc+31
        v16bf ap1 = load_frag_rm(pbuf + 32, 64, lane);  // keys kc+32..kc+63

        // O += P @ V (8 WMMAs, V-fragments already resident)
#pragma unroll
        for (int j = 0; j < 4; ++j) {
            acc[j] = wmma_bf16(ap0, bv[2 * j], acc[j]);
            acc[j] = wmma_bf16(ap1, bv[2 * j + 1], acc[j]);
        }
    }

    float* obase = O + ((size_t)(bo * Tc + qrow0)) * Dc + h * HWc;
#pragma unroll
    for (int i = 0; i < 8; ++i) {
        int row = i + 8 * hi;
        float inv = 1.0f / rsum[i];
#pragma unroll
        for (int j = 0; j < 4; ++j)
            obase[(size_t)row * Dc + j * 16 + ln] = acc[j][i] * inv;
    }
}

// ---------------------------------------------------------------------------
// Host-side orchestration
// ---------------------------------------------------------------------------
extern "C" void kernel_launch(void* const* d_in, const int* in_sizes, int n_in,
                              void* d_out, int out_size, void* d_ws, size_t ws_size,
                              hipStream_t stream) {
    const int*   tokens = (const int*)d_in[0];
    const float* embed  = (const float*)d_in[1];
    const float* Wq = (const float*)d_in[2];
    const float* Wk = (const float*)d_in[3];
    const float* Wv = (const float*)d_in[4];
    const float* W1 = (const float*)d_in[5];
    const float* W2 = (const float*)d_in[6];
    const float* g1 = (const float*)d_in[7];
    const float* b1 = (const float*)d_in[8];
    const float* g2 = (const float*)d_in[9];
    const float* b2 = (const float*)d_in[10];

    char* ws = (char*)d_ws;
    float* x    = (float*)ws;  ws += (size_t)Rc * Dc * 4;
    float* xn_f = (float*)ws;  ws += (size_t)Rc * Dc * 4;
    float* o_f  = (float*)ws;  ws += (size_t)Rc * Dc * 4;
    __bf16* xn_b = (__bf16*)ws; ws += (size_t)Rc * Dc * 2;
    __bf16* q_b  = (__bf16*)ws; ws += (size_t)Rc * Dc * 2;
    __bf16* k_b  = (__bf16*)ws; ws += (size_t)Rc * Dc * 2;
    __bf16* vt_b = (__bf16*)ws; ws += (size_t)Rc * Dc * 2;  // V stored transposed
    __bf16* h_b  = (__bf16*)ws; ws += (size_t)Rc * Dc * 2;
    __bf16* m1_b = (__bf16*)ws; ws += (size_t)Rc * HDc * 2;
    // bf16 weight staging (per layer) + embed (once)
    __bf16* wq_b = (__bf16*)ws; ws += (size_t)Dc * Dc * 2;
    __bf16* wk_b = (__bf16*)ws; ws += (size_t)Dc * Dc * 2;
    __bf16* wv_b = (__bf16*)ws; ws += (size_t)Dc * Dc * 2;
    __bf16* w1_b = (__bf16*)ws; ws += (size_t)HDc * Dc * 2;
    __bf16* w2_b = (__bf16*)ws; ws += (size_t)Dc * HDc * 2;
    __bf16* emb_b = (__bf16*)ws; ws += (size_t)Vc * Dc * 2;

    const int ew_blocks = (Rc * Dc) / 256;
    const int wqkv_blocks = (Dc * Dc) / 256;
    const int wmlp_blocks = (HDc * Dc) / 256;

    gather_kernel<<<ew_blocks, 256, 0, stream>>>(tokens, embed, x);

    dim3 gqkv(Rc / 128, Dc / 256);    // 32 x 2
    dim3 gmlp1(Rc / 128, HDc / 256);  // 32 x 8

    for (int l = 0; l < Lc; ++l) {
        cvt_kernel<<<wqkv_blocks, 256, 0, stream>>>(Wq + (size_t)l * Dc * Dc, wq_b);
        cvt_kernel<<<wqkv_blocks, 256, 0, stream>>>(Wk + (size_t)l * Dc * Dc, wk_b);
        cvt_kernel<<<wqkv_blocks, 256, 0, stream>>>(Wv + (size_t)l * Dc * Dc, wv_b);
        cvt_kernel<<<wmlp_blocks, 256, 0, stream>>>(W1 + (size_t)l * HDc * Dc, w1_b);
        cvt_kernel<<<wmlp_blocks, 256, 0, stream>>>(W2 + (size_t)l * Dc * HDc, w2_b);

        ln_kernel<<<Rc, 256, 0, stream>>>(x, g1 + (size_t)l * Dc, b1 + (size_t)l * Dc, xn_f, xn_b);

        gemm_kernel<1><<<gqkv, 256, 0, stream>>>(xn_b, wq_b, nullptr, nullptr, q_b, Rc, Dc, Dc);
        gemm_kernel<1><<<gqkv, 256, 0, stream>>>(xn_b, wk_b, nullptr, nullptr, k_b, Rc, Dc, Dc);
        gemm_kernel<4><<<gqkv, 256, 0, stream>>>(xn_b, wv_b, nullptr, nullptr, vt_b, Rc, Dc, Dc);

        attn_kernel<<<Bc * Hc * (Tc / 16), 32, 0, stream>>>(q_b, k_b, vt_b, o_f);

        add_kernel<<<ew_blocks, 256, 0, stream>>>(xn_f, o_f, x);  // att_res = xn + o

        ln_kernel<<<Rc, 256, 0, stream>>>(x, g2 + (size_t)l * Dc, b2 + (size_t)l * Dc, xn_f, h_b);

        gemm_kernel<3><<<gmlp1, 256, 0, stream>>>(h_b, w1_b, nullptr, nullptr, m1_b, Rc, HDc, Dc);
        gemm_kernel<2><<<gqkv, 256, 0, stream>>>(m1_b, w2_b, x, x, nullptr, Rc, Dc, HDc);
    }

    // Tied output projection: logits = x @ embed^T
    cvt_kernel<<<ew_blocks, 256, 0, stream>>>(x, xn_b);
    cvt_kernel<<<(Vc * Dc) / 256, 256, 0, stream>>>(embed, emb_b);
    dim3 glog(Rc / 128, Vc / 256);  // 32 x 125
    gemm_kernel<0><<<glog, 256, 0, stream>>>(xn_b, emb_b, nullptr, (float*)d_out, nullptr,
                                             Rc, Vc, Dc);
}
